// PseudoTripletLoss_68513318306546
// MI455X (gfx1250) — matrix operand order repaired and checked by reading, not compiled, not run on hardware
//
#include <hip/hip_runtime.h>

typedef __attribute__((ext_vector_type(16))) _Float16 v16h;
typedef __attribute__((ext_vector_type(8)))  _Float16 v8h;
typedef __attribute__((ext_vector_type(4)))  _Float16 v4h;
typedef __attribute__((ext_vector_type(8)))  float    v8f;

#define N_EMB 8192
#define D_EMB 128
#define NT    512                      // 16x16 tiles per dimension
#define T_TRI (NT * (NT + 1) / 2)      // 131328 upper-tri tiles (incl. diagonal)
#define WAVES_PER_BLOCK 8
#define MARGIN 0.1f

// ---------------------------------------------------------------------------
// Kernel 1: row-normalize fp32 embeddings -> f16 matrix in workspace.
// One wave per row: 4 floats/lane, shfl reduction for the norm.
// ---------------------------------------------------------------------------
__global__ void ptl_normalize(const float* __restrict__ emb,
                              _Float16* __restrict__ en) {
    const int row  = blockIdx.x * WAVES_PER_BLOCK + (threadIdx.x >> 5);
    const int lane = threadIdx.x & 31;
    const float4 x = *(const float4*)(emb + (size_t)row * D_EMB + lane * 4);
    float ss = x.x * x.x + x.y * x.y + x.z * x.z + x.w * x.w;
    #pragma unroll
    for (int off = 16; off; off >>= 1) ss += __shfl_xor(ss, off, 32);
    const float inv = 1.0f / fmaxf(sqrtf(ss), 1e-8f);
    v4h h;
    h[0] = (_Float16)(x.x * inv);
    h[1] = (_Float16)(x.y * inv);
    h[2] = (_Float16)(x.z * inv);
    h[3] = (_Float16)(x.w * inv);
    *(v4h*)(en + (size_t)row * D_EMB + lane * 4) = h;
}

// ---------------------------------------------------------------------------
// Kernel 2: one 16x16 sim tile per wave via v_wmma_f32_16x16x32_f16 (K=128 in
// 4 chunks), fused masked pair-loss, block-level deterministic partial sum.
// ---------------------------------------------------------------------------
__global__ void ptl_tiles(const _Float16* __restrict__ en,
                          const int* __restrict__ labels,
                          float* __restrict__ partials) {
    __shared__ float wsum[WAVES_PER_BLOCK];
    const int wave = threadIdx.x >> 5;
    const int lane = threadIdx.x & 31;
    const long t   = (long)blockIdx.x * WAVES_PER_BLOCK + wave;

    float mySum = 0.0f;
    if (t < (long)T_TRI) {                      // wave-uniform: EXEC stays full
        // Reverse triangular mapping: t -> (ti, tj) with ti <= tj.
        const long u = (long)T_TRI - 1 - t;
        int k = (int)((sqrtf(8.0f * (float)u + 1.0f) - 1.0f) * 0.5f);
        while ((long)(k + 1) * (k + 2) / 2 <= u) ++k;
        while ((long)k * (k + 1) / 2 > u) --k;
        const int  ti = NT - 1 - k;
        const long o  = u - (long)k * (k + 1) / 2;
        const int  tj = NT - 1 - (int)o;

        const int m    = lane & 15;
        const int half = lane >> 4;
        const _Float16* rowA = en + (size_t)(ti * 16 + m) * D_EMB;
        const _Float16* rowB = en + (size_t)(tj * 16 + m) * D_EMB;

        v8f acc = {};
        #pragma unroll
        for (int c = 0; c < 4; ++c) {
            const int kb = c * 32;
            v16h a, b;
            // A 16x32 f16 layout: lane half 0 -> K kb..kb+15,
            //                     lane half 1 -> K kb+16..kb+31
            a = *(const v16h*)(rowA + kb + half * 16);
            // B 32x16: column n = lane&15 is row (tj*16+n) of en; lanes 0-15
            // carry K kb..kb+15, lanes 16-31 carry K kb+16..kb+31.
            b = *(const v16h*)(rowB + kb + half * 16);
            acc = __builtin_amdgcn_wmma_f32_16x16x32_f16(
                false, a, false, b, (short)0, acc, false, false);
        }

        // C layout: lane L holds N = L&15; VGPR v holds M = v + 8*(L>>4).
        const int jCol   = tj * 16 + (lane & 15);
        const int labCol = labels[jCol];
        #pragma unroll
        for (int v = 0; v < 8; ++v) {
            const int   iRow = ti * 16 + v + half * 8;
            const float s    = acc[v];
            const float p    = (labels[iRow] == labCol)
                                   ? fmaxf(1.0f - s, 0.0f)
                                   : fmaxf(s - MARGIN, 0.0f);
            mySum += (iRow < jCol) ? p : 0.0f;   // strict upper triangle
        }
    }

    #pragma unroll
    for (int off = 16; off; off >>= 1) mySum += __shfl_xor(mySum, off, 32);
    if (lane == 0) wsum[wave] = mySum;
    __syncthreads();
    if (threadIdx.x == 0) {
        float s = 0.0f;
        #pragma unroll
        for (int w = 0; w < WAVES_PER_BLOCK; ++w) s += wsum[w];
        partials[blockIdx.x] = s;
    }
}

// ---------------------------------------------------------------------------
// Kernel 3: deterministic single-block reduction of block partials + scaling.
// ---------------------------------------------------------------------------
__global__ void ptl_reduce(const float* __restrict__ partials, int np,
                           float* __restrict__ out) {
    __shared__ float sm[256];
    float s = 0.0f;
    for (int i = threadIdx.x; i < np; i += 256) s += partials[i];
    sm[threadIdx.x] = s;
    __syncthreads();
    #pragma unroll
    for (int off = 128; off; off >>= 1) {
        if ((int)threadIdx.x < off) sm[threadIdx.x] += sm[threadIdx.x + off];
        __syncthreads();
    }
    if (threadIdx.x == 0)
        out[0] = sm[0] * (float)(1.0 / ((double)N_EMB * (double)(N_EMB - 1)));
}

extern "C" void kernel_launch(void* const* d_in, const int* in_sizes, int n_in,
                              void* d_out, int out_size, void* d_ws, size_t ws_size,
                              hipStream_t stream) {
    const float* emb    = (const float*)d_in[0];
    const int*   labels = (const int*)d_in[1];
    float*       out    = (float*)d_out;

    _Float16* en       = (_Float16*)d_ws;                                   // 2 MB
    float*    partials = (float*)((char*)d_ws + (size_t)N_EMB * D_EMB * 2); // 64 KB

    const int tileBlocks = (T_TRI + WAVES_PER_BLOCK - 1) / WAVES_PER_BLOCK; // 16416

    ptl_normalize<<<N_EMB / WAVES_PER_BLOCK, 256, 0, stream>>>(emb, en);
    ptl_tiles<<<tileBlocks, 256, 0, stream>>>(en, labels, partials);
    ptl_reduce<<<1, 256, 0, stream>>>(partials, tileBlocks, out);
}
